// tf_ETE_58583353917526
// MI455X (gfx1250) — matrix operand order repaired and checked by reading, not compiled, not run on hardware
//
#include <hip/hip_runtime.h>

#define NDIM 4096
#define BM 128
#define BN 128
#define BK 32
#define LDSS 40   // 32 + 8 bf16 pad; 80 bytes/row, 16B aligned

typedef __attribute__((ext_vector_type(16))) __bf16 bf16x16;
typedef __attribute__((ext_vector_type(8)))  __bf16 bf16x8;
typedef __attribute__((ext_vector_type(8)))  float  f32x8;
typedef __attribute__((ext_vector_type(4)))  float  f32x4;

// ---------------- rank-1 structure: rowsum / colsum of A ----------------

__global__ __launch_bounds__(256) void rowsum_kernel(const float* __restrict__ A,
                                                     float* __restrict__ rowsum) {
  __shared__ float red[8];
  const int i = blockIdx.x;
  const float* arow = A + (size_t)i * NDIM;
  float s = 0.f;
  for (int k = threadIdx.x; k < NDIM; k += 256) s += arow[k];
  for (int off = 16; off > 0; off >>= 1) s += __shfl_down(s, off, 32);
  if ((threadIdx.x & 31) == 0) red[threadIdx.x >> 5] = s;
  __syncthreads();
  if (threadIdx.x == 0) {
    float t = 0.f;
    for (int w = 0; w < 8; ++w) t += red[w];
    rowsum[i] = t;
  }
}

__global__ __launch_bounds__(256) void colsum_kernel(const float* __restrict__ A,
                                                     float* __restrict__ colsum) {
  const int j = blockIdx.x * 256 + threadIdx.x;
  float s0 = 0.f, s1 = 0.f, s2 = 0.f, s3 = 0.f;
  for (int i = 0; i < NDIM; i += 4) {
    s0 += A[(size_t)(i + 0) * NDIM + j];
    s1 += A[(size_t)(i + 1) * NDIM + j];
    s2 += A[(size_t)(i + 2) * NDIM + j];
    s3 += A[(size_t)(i + 3) * NDIM + j];
  }
  colsum[j] = (s0 + s1) + (s2 + s3);
}

// ------- fused F-generation + bf16x3 split WMMA GEMM, double-buffered -------
// out = F @ W + b,  F[i,k] = (rowsum[i] + colsum[k] - 2 A[i,k]) * A[i,k]
// Block tile 128x128, 8 wave32s in 4(M)x2(N); wave tile 32(M)x64(N).

__global__ __launch_bounds__(256) void fused_gemm_kernel(
    const float* __restrict__ A, const float* __restrict__ W,
    const float* __restrict__ bias, const float* __restrict__ rowsum,
    const float* __restrict__ colsum, float* __restrict__ out)
{
  // Double-buffered operand planes.
  // A planes: [BM][BK] m-major; B planes: [BN][BK] k-major (transposed W tile).
  __shared__ __bf16 lhsHi[2][BM * LDSS];
  __shared__ __bf16 lhsLo[2][BM * LDSS];
  __shared__ __bf16 rhsHi[2][BN * LDSS];
  __shared__ __bf16 rhsLo[2][BN * LDSS];

  const int tid  = threadIdx.x;
  const int lane = tid & 31;
  const int wave = tid >> 5;
  const int m0 = blockIdx.y * BM;
  const int n0 = blockIdx.x * BN;

  const int wm = (wave & 3) * 32;   // wave M offset: two 16-row tiles
  const int wn = (wave >> 2) * 64;  // wave N offset: four 16-col tiles

  const int lrow   = lane & 15;
  const int hihalf = (lane >> 4);          // lanes 0-15 vs 16-31
  const int kA0 = hihalf * 16;             // A frag chunk0 K base (0 / 16)
  const int kA1 = hihalf * 16 + 8;         // A frag chunk1 K base (8 / 24)
  const int kB  = hihalf * 16;             // B frag K base (0 / 16)

  // Staging geometry: 256 threads x 4 row-groups cover each 4096-elem tile.
  const int arow = tid >> 3;          // 0..31  (+32 per group)
  const int acol = (tid & 7) * 4;     // 0..28  (group-invariant)
  const int wrow = tid >> 5;          // 0..7   (+8 per group)
  const int wcol = (tid & 31) * 4;    // 0..124 (group-invariant)

  // ---- prefetch packet: raw f32 tiles + rank-1 vectors, in registers ----
  auto fetch = [&](int k0, f32x4* a, f32x4* w, float* rs, f32x4& cs) {
    #pragma unroll
    for (int r = 0; r < 4; ++r) {
      a[r]  = *(const f32x4*)(A + (size_t)(m0 + arow + 32 * r) * NDIM + k0 + acol);
      w[r]  = *(const f32x4*)(W + (size_t)(k0 + wrow + 8 * r)  * NDIM + n0 + wcol);
      rs[r] = rowsum[m0 + arow + 32 * r];
    }
    cs = *(const f32x4*)(colsum + k0 + acol);
  };

  // ---- transform + bf16 hi/lo split + LDS store into buffer `buf` ----
  auto stage = [&](int buf, const f32x4* a, const f32x4* w,
                   const float* rs, const f32x4& cs) {
    #pragma unroll
    for (int r = 0; r < 4; ++r) {
      const int ar = arow + 32 * r;
      const int wr = wrow + 8 * r;
      #pragma unroll
      for (int q = 0; q < 4; ++q) {
        float av = a[r][q];
        float f  = (rs[r] + cs[q] - 2.0f * av) * av;
        __bf16 h = (__bf16)f;
        __bf16 l = (__bf16)(f - (float)h);
        lhsHi[buf][ar * LDSS + acol + q] = h;
        lhsLo[buf][ar * LDSS + acol + q] = l;

        float wv = w[r][q];
        __bf16 wh = (__bf16)wv;
        __bf16 wl = (__bf16)(wv - (float)wh);
        rhsHi[buf][(wcol + q) * LDSS + wr] = wh;
        rhsLo[buf][(wcol + q) * LDSS + wr] = wl;
      }
    }
  };

  f32x8 acc[2][4] = {};

  // Prologue: stage tile 0.
  {
    f32x4 a[4], w[4], cs; float rs[4];
    fetch(0, a, w, rs, cs);
    stage(0, a, w, rs, cs);
  }
  __syncthreads();

  union Frag { bf16x16 v; bf16x8 h[2]; };

  int buf = 0;
  for (int k0 = 0; k0 < NDIM; k0 += BK) {
    const bool has_next = (k0 + BK) < NDIM;

    // 1) Issue next tile's global loads early; they fly under the WMMAs.
    f32x4 na[4], nw[4], ncs; float nrs[4];
    if (has_next) fetch(k0 + BK, na, nw, nrs, ncs);

    // 2) Build fragments from current buffer and run split-bf16 WMMAs.
    Frag aHi[2], aLo[2], bHi[4], bLo[4];

    #pragma unroll
    for (int mt = 0; mt < 2; ++mt) {
      const __bf16* pAh = &lhsHi[buf][(wm + mt * 16 + lrow) * LDSS];
      const __bf16* pAl = &lhsLo[buf][(wm + mt * 16 + lrow) * LDSS];
      aHi[mt].h[0] = *(const bf16x8*)(pAh + kA0);
      aHi[mt].h[1] = *(const bf16x8*)(pAh + kA1);
      aLo[mt].h[0] = *(const bf16x8*)(pAl + kA0);
      aLo[mt].h[1] = *(const bf16x8*)(pAl + kA1);
    }
    #pragma unroll
    for (int nt = 0; nt < 4; ++nt) {
      const __bf16* pBh = &rhsHi[buf][(wn + nt * 16 + lrow) * LDSS] + kB;
      const __bf16* pBl = &rhsLo[buf][(wn + nt * 16 + lrow) * LDSS] + kB;
      bHi[nt].h[0] = *(const bf16x8*)(pBh);
      bHi[nt].h[1] = *(const bf16x8*)(pBh + 8);
      bLo[nt].h[0] = *(const bf16x8*)(pBl);
      bLo[nt].h[1] = *(const bf16x8*)(pBl + 8);
    }

    #pragma unroll
    for (int mt = 0; mt < 2; ++mt) {
      #pragma unroll
      for (int nt = 0; nt < 4; ++nt) {
        acc[mt][nt] = __builtin_amdgcn_wmma_f32_16x16x32_bf16(
            false, aHi[mt].v, false, bHi[nt].v, (short)0, acc[mt][nt], false, false);
        acc[mt][nt] = __builtin_amdgcn_wmma_f32_16x16x32_bf16(
            false, aHi[mt].v, false, bLo[nt].v, (short)0, acc[mt][nt], false, false);
        acc[mt][nt] = __builtin_amdgcn_wmma_f32_16x16x32_bf16(
            false, aLo[mt].v, false, bHi[nt].v, (short)0, acc[mt][nt], false, false);
      }
    }

    // 3) Stage next tile into the other buffer; single barrier per iteration.
    if (has_next) stage(buf ^ 1, na, nw, nrs, ncs);
    __syncthreads();
    buf ^= 1;
  }

  // ---- epilogue: C/D layout VGPR v -> M = v (+8 for lanes 16-31), N = lane%16 ----
  const int crow = m0 + wm + hihalf * 8;
  #pragma unroll
  for (int mt = 0; mt < 2; ++mt) {
    #pragma unroll
    for (int nt = 0; nt < 4; ++nt) {
      const int cc = n0 + wn + nt * 16 + lrow;
      const float bv = bias[cc];
      #pragma unroll
      for (int v = 0; v < 8; ++v) {
        out[(size_t)(crow + mt * 16 + v) * NDIM + cc] = acc[mt][nt][v] + bv;
      }
    }
  }
}

// ---------------------------------------------------------------------------

extern "C" void kernel_launch(void* const* d_in, const int* in_sizes, int n_in,
                              void* d_out, int out_size, void* d_ws, size_t ws_size,
                              hipStream_t stream) {
  const float* A    = (const float*)d_in[0];
  const float* W    = (const float*)d_in[1];
  const float* bias = (const float*)d_in[2];
  float* out = (float*)d_out;

  float* rowsum = (float*)d_ws;        // NDIM floats
  float* colsum = rowsum + NDIM;       // NDIM floats

  rowsum_kernel<<<NDIM, 256, 0, stream>>>(A, rowsum);
  colsum_kernel<<<NDIM / 256, 256, 0, stream>>>(A, colsum);

  dim3 grid(NDIM / BN, NDIM / BM);
  fused_gemm_kernel<<<grid, 256, 0, stream>>>(A, W, bias, rowsum, colsum, out);
}
